// SparseMoEBlock_87806311399468
// MI455X (gfx1250) — matrix operand order repaired and checked by reading, not compiled, not run on hardware
//
#include <hip/hip_runtime.h>
#include <math.h>

// ---- CDNA5 vector types for WMMA / async DMA ----
typedef __attribute__((ext_vector_type(16))) __bf16 v16bf;
typedef __attribute__((ext_vector_type(8)))  __bf16 v8bf;
typedef __attribute__((ext_vector_type(8)))  float  v8f;
typedef __attribute__((ext_vector_type(4)))  int    v4i;

// Async global->LDS path (gfx1250). Fallback keeps the synchronous copy so the
// build never regresses if the toolchain lacks the builtin.
#if defined(__gfx1250__) && __has_builtin(__builtin_amdgcn_global_load_async_to_lds_b128) && __has_builtin(__builtin_amdgcn_s_wait_asynccnt)
#define MOE_ASYNC 1
#else
#define MOE_ASYNC 0
#endif

// Problem constants (match reference)
constexpr int Bc = 2, Sc = 2048, Dc = 1024, Fc = 4096, Ec = 8, Kc = 2;
constexpr int NT    = Bc * Sc;          // 4096 tokens
constexpr int NSLOT = NT * Kc;          // 8192 token-slots
constexpr int BM = 128, BN = 128, BK = 32;
constexpr int MPAD   = NSLOT + Ec * BM; // 9216 rows max after 128-alignment per expert
constexpr int MTILES = NSLOT / BM + Ec; // 72 M-tiles max

// ------------------------------------------------------------------
// Gate: logits -> softmax -> top-2 (strict > keeps lowest index, like jax)
// ------------------------------------------------------------------
__global__ void moe_gate(const float* __restrict__ x, const float* __restrict__ gw,
                         int* __restrict__ tki, float* __restrict__ tkp,
                         int* __restrict__ counts) {
  int t = blockIdx.x;
  int tid = threadIdx.x, lane = tid & 31, wid = tid >> 5;
  const float* xr = x + (size_t)t * Dc;
  const float* gr = gw + (size_t)wid * Dc;
  float s = 0.f;
  for (int d = lane; d < Dc; d += 32) s += xr[d] * gr[d];
  for (int o = 16; o > 0; o >>= 1) s += __shfl_xor(s, o, 32);
  __shared__ float lg[Ec];
  if (lane == 0) lg[wid] = s;
  __syncthreads();
  if (tid == 0) {
    float m = lg[0];
    for (int e = 1; e < Ec; e++) m = fmaxf(m, lg[e]);
    float p[Ec], sum = 0.f;
    for (int e = 0; e < Ec; e++) { p[e] = expf(lg[e] - m); sum += p[e]; }
    float inv = 1.f / sum;
    for (int e = 0; e < Ec; e++) p[e] *= inv;
    int e0 = 0;
    for (int e = 1; e < Ec; e++) if (p[e] > p[e0]) e0 = e;
    int e1 = (e0 == 0) ? 1 : 0;
    for (int e = 0; e < Ec; e++) if (e != e0 && p[e] > p[e1]) e1 = e;
    tki[t * 2 + 0] = e0; tki[t * 2 + 1] = e1;
    tkp[t * 2 + 0] = p[e0]; tkp[t * 2 + 1] = p[e1];
    atomicAdd(&counts[e0], 1);
    atomicAdd(&counts[e1], 1);
  }
}

__global__ void moe_zero(int* __restrict__ counts) {
  if (threadIdx.x < Ec) counts[threadIdx.x] = 0;
}

// Per-expert segment offsets, 128-row aligned so GEMM tiles never straddle experts.
__global__ void moe_scan(const int* __restrict__ counts, int* __restrict__ offp,
                         int* __restrict__ cur) {
  if (threadIdx.x == 0) {
    int a = 0;
    for (int e = 0; e < Ec; e++) {
      offp[e] = a; cur[e] = a;
      a += (counts[e] + BM - 1) & ~(BM - 1);
    }
    offp[Ec] = a;
  }
}

// Gather token rows into expert-grouped bf16 buffer.
__global__ void moe_gather(const float* __restrict__ x, const int* __restrict__ tki,
                           int* __restrict__ cur, int* __restrict__ pos_of_slot,
                           __bf16* __restrict__ xg) {
  int s = blockIdx.x, t = s >> 1;
  __shared__ int sp;
  if (threadIdx.x == 0) {
    int e = tki[s];
    int p = atomicAdd(&cur[e], 1);
    pos_of_slot[s] = p;
    sp = p;
  }
  __syncthreads();
  int pos = sp;
  const float* xr = x + (size_t)t * Dc;
  __bf16* dst = xg + (size_t)pos * Dc;
  for (int d = threadIdx.x; d < Dc; d += blockDim.x) dst[d] = (__bf16)xr[d];
}

// fp32 -> bf16 weight conversion (one pass; halves weight traffic afterwards)
__global__ void cvt_bf16(const float* __restrict__ src, __bf16* __restrict__ dst, long n) {
  long i = (long)blockIdx.x * blockDim.x + threadIdx.x;
  long stride = (long)gridDim.x * blockDim.x;
  for (; i < n; i += stride) dst[i] = (__bf16)src[i];
}

__device__ __forceinline__ float gelu_tanh(float v) {
  float c = v + 0.044715f * v * v * v;
  return 0.5f * v * (1.f + tanhf(0.7978845608028654f * c));
}

// 16B global -> LDS copy: async DMA on CDNA5, VGPR round-trip otherwise.
__device__ __forceinline__ void copy16_g2l(__bf16* dst_lds, const __bf16* src_g) {
#if MOE_ASYNC
  __builtin_amdgcn_global_load_async_to_lds_b128(
      (__attribute__((address_space(1))) v4i*)src_g,
      (__attribute__((address_space(3))) v4i*)dst_lds, 0, 0);
#else
  *(v8bf*)dst_lds = *(const v8bf*)src_g;
#endif
}

// ------------------------------------------------------------------
// Grouped GEMM: C[m,n] = sum_k A[m,k] * Bw[n*ldbn + e*K + k]  (+ bias[e,n])
// A: bf16 row-major (lda == K). Bw: "n-major, k-contiguous" bf16 (both w1,w2 fit).
// LAYER1: epilogue = +b1, GELU, bf16 store. LAYER2: epilogue = +b2, f32 store.
// 256 threads = 8 waves in a 2x4 grid; each wave owns 64x32 (8 WMMA tiles).
// Double-buffered LDS; tile k+1 streams in via ASYNCcnt while tile k computes.
// ------------------------------------------------------------------
template <bool LAYER1>
__global__ __launch_bounds__(256)
void moe_gemm(const __bf16* __restrict__ A, const __bf16* __restrict__ Bw,
              const float* __restrict__ bias, __bf16* __restrict__ O16,
              float* __restrict__ O32, const int* __restrict__ offp,
              int K, int ldbn, int N) {
  __shared__ __align__(32) __bf16 sA[2][BM * BK]; // [buf][m][k]
  __shared__ __align__(32) __bf16 sB[2][BN * BK]; // [buf][n][k]

  int m0 = blockIdx.x * BM;
  int total = offp[Ec];
  if (m0 >= total) return;
  int e = 0;
  while (e < Ec - 1 && m0 >= offp[e + 1]) e++;
  int n0 = blockIdx.y * BN;
  long eoff = (long)e * K;

  int tid = threadIdx.x, lane = tid & 31, wid = tid >> 5;
  int wm = wid >> 2, wn = wid & 3;        // 2 x 4 wave grid
  int lrow = lane & 15, lhi = lane >> 4;  // ISA lane split (lanes 0-15 / 16-31)

  // Per-thread staging of one k-tile: 4 x 16B chunks (2 for A, 2 for B).
  auto stage = [&](int buf, int k0) {
#pragma unroll
    for (int tch = 0; tch < 2; tch++) {
      int ch = tid + tch * 256;
      int r = ch >> 2, kc = (ch & 3) * 8;
      copy16_g2l(&sA[buf][r * BK + kc],
                 &A[(size_t)(m0 + r) * K + k0 + kc]);
      copy16_g2l(&sB[buf][r * BK + kc],
                 &Bw[(size_t)(n0 + r) * (size_t)ldbn + eoff + k0 + kc]);
    }
  };

  v8f acc[4][2];
#pragma unroll
  for (int i = 0; i < 4; i++)
#pragma unroll
    for (int j = 0; j < 2; j++) acc[i][j] = {};

  stage(0, 0);  // prologue: tile 0 in flight
  int buf = 0;
  for (int k0 = 0; k0 < K; k0 += BK) {
    int nxt = buf ^ 1;
    bool has_next = (k0 + BK < K);
    __syncthreads();          // all waves done reading buf[nxt] from prior iter
    if (has_next) stage(nxt, k0 + BK);
    if (k0 + 2 * BK < K) {    // warm caches two tiles ahead
      int r = tid & 127;
      __builtin_prefetch(&A[(size_t)(m0 + r) * K + k0 + 2 * BK], 0, 3);
      __builtin_prefetch(&Bw[(size_t)(n0 + r) * (size_t)ldbn + eoff + k0 + 2 * BK], 0, 3);
    }
#if MOE_ASYNC
    // Async loads complete in order: waiting <=4 leaves only tile k+1 in flight.
    if (has_next) __builtin_amdgcn_s_wait_asynccnt(4);
    else          __builtin_amdgcn_s_wait_asynccnt(0);
#endif
    __syncthreads();          // buf[buf] fully resident for every wave

    // Build fragments: A and B both in contiguous k order (k = 16*lhi + element).
    v16bf afr[4], bfr[2];
#pragma unroll
    for (int i = 0; i < 4; i++) {
      // A 16x32: lanes 0-15 row M=lrow K[0..15]; lanes 16-31 K[16..31]
      const __bf16* p = &sA[buf][(wm * 64 + i * 16 + lrow) * BK + lhi * 16];
      v8bf lo = *(const v8bf*)p;
      v8bf hi = *(const v8bf*)(p + 8);
      afr[i] = __builtin_shufflevector(lo, hi, 0, 1, 2, 3, 4, 5, 6, 7,
                                       8, 9, 10, 11, 12, 13, 14, 15);
    }
#pragma unroll
    for (int j = 0; j < 2; j++) {
      // B 32x16: lane holds column n=lrow, K=lhi*16 .. +15 contiguous
      const __bf16* p = &sB[buf][(wn * 32 + j * 16 + lrow) * BK + lhi * 16];
      bfr[j] = *(const v16bf*)p;
    }
#pragma unroll
    for (int i = 0; i < 4; i++)
#pragma unroll
      for (int j = 0; j < 2; j++)
        acc[i][j] = __builtin_amdgcn_wmma_f32_16x16x32_bf16(
            false, afr[i], false, bfr[j], (short)0, acc[i][j], false, false);
    buf = nxt;
  }

  // Epilogue. C layout: lane(0-15): VGPR r -> M=r; lane(16-31): M=r+8; N = lrow.
#pragma unroll
  for (int i = 0; i < 4; i++) {
#pragma unroll
    for (int j = 0; j < 2; j++) {
      int gn = n0 + wn * 32 + j * 16 + lrow;
      float bv = bias[(size_t)e * N + gn];
#pragma unroll
      for (int r = 0; r < 8; r++) {
        int gm = m0 + wm * 64 + i * 16 + lhi * 8 + r;
        float v = acc[i][j][r] + bv;
        if constexpr (LAYER1) {
          O16[(size_t)gm * N + gn] = (__bf16)gelu_tanh(v);
        } else {
          O32[(size_t)gm * N + gn] = v;
        }
      }
    }
  }
}

// out[t] = p0 * y[pos0] + p1 * y[pos1]
__global__ void moe_combine(const float* __restrict__ y, const int* __restrict__ pos_of_slot,
                            const float* __restrict__ tkp, float* __restrict__ out) {
  int t = blockIdx.x;
  int p0 = pos_of_slot[t * 2 + 0], p1 = pos_of_slot[t * 2 + 1];
  float w0 = tkp[t * 2 + 0], w1 = tkp[t * 2 + 1];
  const float* y0 = y + (size_t)p0 * Dc;
  const float* y1 = y + (size_t)p1 * Dc;
  float* o = out + (size_t)t * Dc;
  for (int d = threadIdx.x; d < Dc; d += blockDim.x)
    o[d] = w0 * y0[d] + w1 * y1[d];
}

extern "C" void kernel_launch(void* const* d_in, const int* in_sizes, int n_in,
                              void* d_out, int out_size, void* d_ws, size_t ws_size,
                              hipStream_t stream) {
  (void)in_sizes; (void)n_in; (void)out_size; (void)ws_size;
  const float* x  = (const float*)d_in[0];
  const float* gw = (const float*)d_in[1];
  const float* w1 = (const float*)d_in[2];
  const float* b1 = (const float*)d_in[3];
  const float* w2 = (const float*)d_in[4];
  const float* b2 = (const float*)d_in[5];
  float* out = (float*)d_out;

  char* ws = (char*)d_ws;
  size_t off = 0;
  auto alloc = [&](size_t bytes) -> void* {
    void* p = ws + off;
    off = (off + bytes + 255) & ~(size_t)255;
    return p;
  };
  int*    counts = (int*)alloc(Ec * 4);
  int*    cur    = (int*)alloc(Ec * 4);
  int*    offp   = (int*)alloc((Ec + 1) * 4);
  int*    tki    = (int*)alloc((size_t)NSLOT * 4);
  float*  tkp    = (float*)alloc((size_t)NSLOT * 4);
  int*    pos    = (int*)alloc((size_t)NSLOT * 4);
  __bf16* xg     = (__bf16*)alloc((size_t)MPAD * Dc * 2);
  __bf16* h      = (__bf16*)alloc((size_t)MPAD * Fc * 2);
  float*  y      = (float*)alloc((size_t)MPAD * Dc * 4);
  __bf16* w1b    = (__bf16*)alloc((size_t)Fc * Ec * Dc * 2);
  __bf16* w2b    = (__bf16*)alloc((size_t)Dc * Ec * Fc * 2);

  moe_zero<<<1, 32, 0, stream>>>(counts);
  cvt_bf16<<<2048, 256, 0, stream>>>(w1, w1b, (long)Fc * Ec * Dc);
  cvt_bf16<<<2048, 256, 0, stream>>>(w2, w2b, (long)Dc * Ec * Fc);
  moe_gate<<<NT, 256, 0, stream>>>(x, gw, tki, tkp, counts);
  moe_scan<<<1, 32, 0, stream>>>(counts, offp, cur);
  moe_gather<<<NSLOT, 256, 0, stream>>>(x, tki, cur, pos, xg);
  moe_gemm<true><<<dim3(MTILES, Fc / BN), 256, 0, stream>>>(
      xg, w1b, b1, h, nullptr, offp, Dc, Ec * Dc, Fc);
  moe_gemm<false><<<dim3(MTILES, Dc / BN), 256, 0, stream>>>(
      h, w2b, b2, nullptr, y, offp, Fc, Ec * Fc, Dc);
  moe_combine<<<NT, 256, 0, stream>>>(y, pos, tkp, out);
}
